// RowLSTM_81913616270013
// MI455X (gfx1250) — matrix operand order repaired and checked by reading, not compiled, not run on hardware
//
#include <hip/hip_runtime.h>

// ---------------------------------------------------------------------------
// RowLSTM fused kernel for MI455X (gfx1250, wave32, WMMA).
//
// One fused GEMM per row: K = 256 (input-to-state, taps {w-1,w}) +
// 384 (state-to-state, taps {w-1,w,w+1}) = 640 = 20 x WMMA_F32_16X16X32_BF16.
// K is ordered TAP-MAJOR (K = t*128 + i), so for each 32-wide K-step the
// source panel (x vs h), tap t and channel base are compile-time constants
// after full unroll: the B fragment is two contiguous ds_load_b128 from a
// transposed LDS panel. 16 workgroups (one batch each), 1024 threads =
// 32 waves = 8 channel-blocks x 4 N-tiles; each wave owns the SAME 16 hid
// channels for all four gates, so the LSTM update is pure register math and
// the cell state never leaves VGPRs.
//
// The A (weight) panel is re-streamed from global every row via an opaque
// addrspace(1) pointer: opacity blocks LICM (which otherwise hoists all 80
// fragments and spills to scratch); the explicit global address space keeps
// the loads on the global_load path (flat loads would also tick DScnt and
// contend with the LDS pipe feeding the B fragments).
// ---------------------------------------------------------------------------

typedef __attribute__((ext_vector_type(16))) __bf16 v16bf;
typedef __attribute__((ext_vector_type(8)))  __bf16 v8bf;
typedef __attribute__((ext_vector_type(8)))  float  v8f;
typedef const v16bf __attribute__((address_space(1)))* gv16bf_p;
typedef const v8bf  __attribute__((address_space(1)))* gv8bf_p;

#define HID    128
#define KER    3
#define NB     16
#define NH     64
#define NW     64
#define KSTEPS 20           // 640 / 32
#define MTILES 32           // 512 / 16
#define LROW   136          // 128 ch + 8 pad (272 B row: 16B aligned, 68-dword stride)

// d_ws layout:
//   [0, 655360)      : A bf16 [Mtile=32][Kstep=20][lane=32][16] (WMMA A-frag layout)
//   [655360, 657408) : bias f32 [512] = b_is + b_ss
#define WS_A_BYTES (MTILES * KSTEPS * 32 * 16 * 2)

// Hardware tanh transcendental when the toolchain exposes it.
#if __has_builtin(__builtin_amdgcn_tanhf)
__device__ __forceinline__ float fast_tanh(float x) { return __builtin_amdgcn_tanhf(x); }
#elif __has_builtin(__builtin_amdgcn_tanh_f32)
__device__ __forceinline__ float fast_tanh(float x) { return __builtin_amdgcn_tanh_f32(x); }
#else
__device__ __forceinline__ float fast_tanh(float x) { return tanhf(x); }
#endif
// sigmoid(z) == 0.5*tanh(z/2) + 0.5 : one trans op, no divide
__device__ __forceinline__ float fast_sigmoid(float z) {
    return __builtin_fmaf(0.5f, fast_tanh(0.5f * z), 0.5f);
}

// ---------------------------------------------------------------------------
// Prep: swizzle (masked W_is | W_ss) into per-lane WMMA A-fragment layout.
// Tap-major K:  K<256 : t=K>>7, i=K&127  -> W_is[o,i,t]   (mask keeps t=0,1)
//               K>=256: kk=K-256, t=kk>>7, i=kk&127 -> W_ss[o,i,t]
// A-matrix 16-bit per-lane layout (contiguous K, matches B fragment):
//   lanes 0-15 : M = lane,    K(j) = j
//   lanes 16-31: M = lane-16, K(j) = j+16
// ---------------------------------------------------------------------------
__global__ void rowlstm_prep(const float* __restrict__ W_is,
                             const float* __restrict__ b_is,
                             const float* __restrict__ W_ss,
                             const float* __restrict__ b_ss,
                             __bf16* __restrict__ A,
                             float* __restrict__ bias) {
    int tid = blockIdx.x * blockDim.x + threadIdx.x;
    if (tid < 4 * HID) bias[tid] = b_is[tid] + b_ss[tid];
    if (tid >= MTILES * KSTEPS * 32 * 16) return;

    int j    = tid & 15;
    int lane = (tid >> 4) & 31;
    int ks   = (tid >> 9) % KSTEPS;
    int m    = (tid >> 9) / KSTEPS;

    int kl = (lane < 16) ? j : j + 16;
    int K  = ks * 32 + kl;
    int o  = m * 16 + (lane & 15);

    float w;
    if (K < 2 * HID) {                          // input-to-state (mask B)
        int t = K >> 7, i = K & 127;
        w = W_is[(o * HID + i) * KER + t];
    } else {                                    // state-to-state
        int kk = K - 2 * HID;
        int t = kk >> 7, i = kk & 127;
        w = W_ss[(o * HID + i) * KER + t];
    }
    A[tid] = (__bf16)w;
}

// ---------------------------------------------------------------------------
// Fused recurrence. Grid = NB blocks (one batch each), 1024 threads.
// ---------------------------------------------------------------------------
__global__ __launch_bounds__(1024)
void rowlstm_scan(const float* __restrict__ x,
                  const __bf16* __restrict__ A,
                  const float* __restrict__ bias,
                  float* __restrict__ out) {
    // Transposed bf16 operand panels: [column + halo][channel].
    // Column c of the image lives in row c+1; rows 0 and 65 are the zero halo.
    __shared__ __align__(16) __bf16 xT[NW + 2][LROW];
    __shared__ __align__(16) __bf16 hT[NW + 2][LROW];

    const int b    = blockIdx.x;
    const int tid  = threadIdx.x;
    const int lane = tid & 31;
    const int wave = tid >> 5;
    const int cb   = wave >> 2;                 // hid-channel block, 0..7
    const int nt   = wave & 3;                  // N (width) tile, 0..3
    const int ncol = nt * 16 + (lane & 15);     // this lane's output column
    const int mh   = (lane < 16) ? 0 : 8;       // C/D layout: M = v + mh
    const int koff = (lane < 16) ? 0 : 16;      // B layout: lanes 16-31 hold K+16

    // Per-wave A base in the GLOBAL address space; (gate, ks) offsets are
    // immediates (gate stride 163840 B, ks stride 1024 B, within 24-bit IOFFSET).
    gv16bf_p Aw = (gv16bf_p)((const v16bf*)A + (size_t)cb * (KSTEPS * 32) + lane);

    // zero h(-1) and the x halo rows
    for (int e = tid; e < (NW + 2) * LROW; e += 1024)
        ((__bf16*)hT)[e] = (__bf16)0.f;
    if (tid < LROW) {
        xT[0][tid]      = (__bf16)0.f;
        xT[NW + 1][tid] = (__bf16)0.f;
    }

    // per-lane biases for the 4 gates (same 16 channels, all gates)
    float bI[8], bF[8], bO[8], bG[8];
#pragma unroll
    for (int v = 0; v < 8; ++v) {
        int ch = cb * 16 + v + mh;
        bI[v] = bias[0 * HID + ch];
        bF[v] = bias[1 * HID + ch];
        bO[v] = bias[2 * HID + ch];
        bG[v] = bias[3 * HID + ch];
    }
    float cst[8] = {};                          // cell state, lives in VGPRs
    __syncthreads();

    for (int r = 0; r < NH; ++r) {
        // stage x row (transposed, bf16) into LDS: 8 elements per thread
        for (int e = tid; e < HID * NW; e += 1024) {
            int i = e >> 6, w = e & 63;
            xT[w + 1][i] = (__bf16)x[((b * HID + i) * NH + r) * NW + w];
        }
        __syncthreads();

        // Opaque per-row copy of the A pointer: blocks LICM from hoisting all
        // 80 fragments out of the row loop (which spills to scratch). A is
        // L2-resident; re-streaming it each row is the intended path.
        gv16bf_p Ar = Aw;
        asm volatile("" : "+v"(Ar));

        v8f accI = {}, accF = {}, accO = {}, accG = {};
#pragma unroll
        for (int ks = 0; ks < KSTEPS; ++ks) {
            // B fragment: 16 contiguous K-channels of panel row (ncol + t).
            // ks 0..7  -> x panel, t = ks>>2,     ibase = (ks&3)*32
            // ks 8..19 -> h panel, t = (ks-8)>>2, ibase = ((ks-8)&3)*32
            const __bf16* src;
            if (ks < 8) {
                src = &xT[ncol + (ks >> 2)][(ks & 3) * 32 + koff];
            } else {
                int kk = ks - 8;
                src = &hT[ncol + (kk >> 2)][(kk & 3) * 32 + koff];
            }
            v8bf lo = *(const v8bf*)(src);
            v8bf hi = *(const v8bf*)(src + 8);
            v16bf bfrag = __builtin_shufflevector(
                lo, hi, 0, 1, 2, 3, 4, 5, 6, 7, 8, 9, 10, 11, 12, 13, 14, 15);

            // 4 gates share the B fragment; A streamed from L2.
            v16bf aI = Ar[0 * 8 * KSTEPS * 32 + ks * 32];
            v16bf aF = Ar[1 * 8 * KSTEPS * 32 + ks * 32];
            v16bf aO = Ar[2 * 8 * KSTEPS * 32 + ks * 32];
            v16bf aG = Ar[3 * 8 * KSTEPS * 32 + ks * 32];
            accI = __builtin_amdgcn_wmma_f32_16x16x32_bf16(
                false, aI, false, bfrag, (short)0, accI, false, false);
            accF = __builtin_amdgcn_wmma_f32_16x16x32_bf16(
                false, aF, false, bfrag, (short)0, accF, false, false);
            accO = __builtin_amdgcn_wmma_f32_16x16x32_bf16(
                false, aO, false, bfrag, (short)0, accO, false, false);
            accG = __builtin_amdgcn_wmma_f32_16x16x32_bf16(
                false, aG, false, bfrag, (short)0, accG, false, false);
            if (ks + 1 < KSTEPS)                // warm L0/GL2 for next K-step
                __builtin_prefetch((const void*)(Ar + (ks + 1) * 32), 0, 3);
        }
        __syncthreads();                        // all hT reads done

        // Gate fusion + LSTM update: fragments are elementwise-aligned.
        v8bf hv;
#pragma unroll
        for (int v = 0; v < 8; ++v) {
            float zi = accI[v] + bI[v];
            float zf = accF[v] + bF[v];
            float zo = accO[v] + bO[v];
            float zg = accG[v] + bG[v];
            float ig = fast_sigmoid(zi);
            float fg = fast_sigmoid(zf);
            float og = fast_sigmoid(zo);
            float gg = fast_tanh(zg);
            float c  = fg * cst[v] + ig * gg;
            cst[v]   = c;
            float h  = og * fast_tanh(c);

            int ch = cb * 16 + v + mh;
            out[((b * HID + ch) * NH + r) * NW + ncol] = h;
            hv[v] = (__bf16)h;
        }
        // 8 contiguous channels -> one 16B LDS store into next row's h panel
        *(v8bf*)&hT[ncol + 1][cb * 16 + mh] = hv;
        __syncthreads();                        // hT ready for next row
    }
}

// ---------------------------------------------------------------------------
extern "C" void kernel_launch(void* const* d_in, const int* in_sizes, int n_in,
                              void* d_out, int out_size, void* d_ws, size_t ws_size,
                              hipStream_t stream) {
    const float* x    = (const float*)d_in[0];
    const float* W_is = (const float*)d_in[1];
    const float* b_is = (const float*)d_in[2];
    const float* W_ss = (const float*)d_in[3];
    const float* b_ss = (const float*)d_in[4];
    float* out = (float*)d_out;

    __bf16* A   = (__bf16*)d_ws;
    float* bias = (float*)((char*)d_ws + WS_A_BYTES);

    int totalA = MTILES * KSTEPS * 32 * 16;
    rowlstm_prep<<<(totalA + 255) / 256, 256, 0, stream>>>(W_is, b_is, W_ss, b_ss,
                                                           A, bias);
    rowlstm_scan<<<NB, 1024, 0, stream>>>(x, A, bias, out);
}
